// SMNN_59828894433308
// MI455X (gfx1250) — compile-verified, matrix operand order checked by reading, not run on hardware
//
#include <hip/hip_runtime.h>

// ---------------------------------------------------------------------------
// Types for CDNA5 WMMA (wave32, 16x16x32 bf16 -> f32)
// ---------------------------------------------------------------------------
typedef __attribute__((ext_vector_type(16))) __bf16 v16bf;
typedef __attribute__((ext_vector_type(8)))  float  v8f;
typedef __attribute__((ext_vector_type(4)))  __bf16 v4bf;
typedef __attribute__((ext_vector_type(4)))  int    v4i;

#define AS1 __attribute__((address_space(1)))
#define AS3 __attribute__((address_space(3)))

#if __has_builtin(__builtin_amdgcn_global_load_async_to_lds_b128) && \
    __has_builtin(__builtin_amdgcn_s_wait_asynccnt)
#define LSTM_ASYNC_X 1
#define LSTM_NBUF 2
#else
#define LSTM_NBUF 1
#endif

__device__ __forceinline__ float sigmoidf_(float x) {
  return 1.0f / (1.0f + __expf(-x));
}
// tanh via exp (v_exp_f32 TRANS op, co-executes with WMMA); clamp avoids inf/nan.
__device__ __forceinline__ float tanh_fast(float x) {
  x = fminf(fmaxf(x, -10.0f), 10.0f);
  float e = __expf(-2.0f * x);
  return (1.0f - e) / (1.0f + e);
}

// ---------------------------------------------------------------------------
// Weight prep: Wcat[n][0..511] = w_ih[n][k], Wcat[n][512..767] = w_hh[n][k-512]
// stored bf16 row-major [1024][768]
// ---------------------------------------------------------------------------
__global__ void build_wcat_kernel(const float* __restrict__ w_ih,
                                  const float* __restrict__ w_hh,
                                  __bf16* __restrict__ out) {
  int id = blockIdx.x * blockDim.x + threadIdx.x;
  if (id >= 1024 * 768) return;
  int n = id / 768, k = id % 768;
  float v = (k < 512) ? w_ih[n * 512 + k] : w_hh[n * 256 + (k - 512)];
  out[id] = (__bf16)v;
}

// ---------------------------------------------------------------------------
// Embedding gather + f32 -> bf16 convert.  One thread per 4 elements.
// ---------------------------------------------------------------------------
__global__ void gather_embed_kernel(const float* __restrict__ emb,
                                    const int* __restrict__ ids,
                                    __bf16* __restrict__ X, long nTok) {
  long i = (long)blockIdx.x * blockDim.x + threadIdx.x;
  long total = nTok * 128;  // 512/4 per token
  if (i >= total) return;
  long tok = i >> 7;
  int part = (int)(i & 127);
  const float4 v = *(const float4*)(emb + (long)ids[tok] * 512 + part * 4);
  v4bf o;
  o.x = (__bf16)v.x; o.y = (__bf16)v.y; o.z = (__bf16)v.z; o.w = (__bf16)v.w;
  *(v4bf*)(X + i * 4) = o;
}

// ---------------------------------------------------------------------------
// Masked LSTM recurrence, one direction.  16 batch rows per workgroup,
// 512 threads = 16 waves.  Wave w owns gate column tiles {w, w+1, w+2, w+3}
// used as i/f/g/o for hidden cols 16w..16w+15, so the cell update is lane-local.
// Per step: gates[16,1024] = [x_t | h] @ Wcat^T via WMMA bf16 (K=768),
// fully unrolled (96 WMMA/step).  The weight pointer is laundered through an
// empty asm each step (as an integer, rebuilt as an addrspace(1) pointer) so
// the loop-invariant B fragments are re-streamed from the shared, L2-resident
// Wcat via global_load_b128 (LOADcnt only) instead of being hoisted + spilled
// to per-wave scratch, and independent of the DScnt-tracked LDS A loads.
// x_t is double-buffered through LDS with async loads (ASYNCcnt) if available.
// ---------------------------------------------------------------------------
__global__ __launch_bounds__(512) void lstm_rec_kernel(
    const __bf16* __restrict__ X,     // [N, T, 512] bf16
    const int*    __restrict__ lens,  // [N]
    const __bf16* __restrict__ Wcat,  // [1024, 768] bf16
    const float*  __restrict__ bias,  // [1024]
    float* __restrict__ out,          // out[(row)*rowStride + colOff + j]
    int T, int rowStride, int colOff, int backward) {
  constexpr int XP = 528;  // padded x row (bf16 elems)
  constexpr int HP = 272;  // padded h row
  __shared__ __bf16 sh_x[LSTM_NBUF * 16 * XP];
  __shared__ __bf16 sh_h[16 * HP];
  __shared__ int sh_len[16];

  const int tid  = threadIdx.x;
  const int wave = tid >> 5;
  const int lane = tid & 31;
  const int l16  = lane & 15;
  const int hi   = lane >> 4;
  const int row0 = blockIdx.x * 16;

  if (tid < 16) sh_len[tid] = lens[row0 + tid];
  for (int i = tid; i < 16 * HP; i += 512) sh_h[i] = (__bf16)0.0f;

  float creg[8], hreg[8];
#pragma unroll
  for (int v = 0; v < 8; ++v) { creg[v] = 0.0f; hreg[v] = 0.0f; }

  const int j = wave * 16 + l16;  // hidden column owned by this lane
  const float bI = bias[j];
  const float bF = bias[256 + j];
  const float bG = bias[512 + j];
  const float bO = bias[768 + j];
  const unsigned long wbits =
      (unsigned long)(Wcat + (long)(wave * 16 + l16) * 768 + hi * 16);

  // stage 16 rows x 512 bf16 of x_t into an LDS buffer (1024 chunks of 16B)
  auto issue_fill = [&](int tstep, __bf16* buf) {
#pragma unroll
    for (int p = 0; p < 2; ++p) {
      int chunk = tid + p * 512;
      int r = chunk >> 6, cid = chunk & 63;
      int len = sh_len[r];
      int tt = backward ? (len - 1 - tstep) : tstep;
      tt = tt < 0 ? 0 : (tt >= T ? T - 1 : tt);
      const void* src = (const void*)(X + (((long)(row0 + r)) * T + tt) * 512 + cid * 8);
      void* dst = (void*)(buf + r * XP + cid * 8);
#ifdef LSTM_ASYNC_X
      __builtin_amdgcn_global_load_async_to_lds_b128(
          (AS1 v4i*)(unsigned long)src,
          (AS3 v4i*)(unsigned int)(unsigned long)dst, 0, 0);
#else
      *(uint4*)dst = *(const uint4*)src;
#endif
    }
  };

  __syncthreads();  // sh_len / sh_h ready
#ifdef LSTM_ASYNC_X
  issue_fill(0, sh_x);
#endif

  for (int t = 0; t < T; ++t) {
#ifdef LSTM_ASYNC_X
    __bf16* cur = sh_x + (t & 1) * (16 * XP);
    __bf16* nxt = sh_x + ((t + 1) & 1) * (16 * XP);
    if (t + 1 < T) {
      issue_fill(t + 1, nxt);                      // overlap with this step
      __builtin_amdgcn_s_wait_asynccnt((short)2);  // fill(t) retired
    } else {
      __builtin_amdgcn_s_wait_asynccnt((short)0);
    }
    __syncthreads();  // all lanes' x_t in LDS + h_{t-1} stores visible
#else
    __bf16* cur = sh_x;
    issue_fill(t, cur);
    __syncthreads();
#endif
    // Launder the weight address so the B-fragment loads stay loop-variant
    // (re-streamed from L2 each step, never hoisted + spilled).  Rebuilt as
    // an addrspace(1) pointer so they lower to global_load_b128 (LOADcnt).
    unsigned long wb = wbits;
    asm volatile("" : "+v"(wb));
    const AS1 __bf16* wr_t = (const AS1 __bf16*)wb;
    __builtin_prefetch((const void*)wb, 0, 3);  // nudge weight rows near

    // -------- gates = [x_t | h] @ Wcat^T  (K = 768 = 24 chunks of 32) ------
    v8f acc[4];
    const v8f zero = {0.f, 0.f, 0.f, 0.f, 0.f, 0.f, 0.f, 0.f};
#pragma unroll
    for (int q = 0; q < 4; ++q) acc[q] = zero;

#pragma unroll
    for (int kc = 0; kc < 16; ++kc) {  // x part
      v16bf a = *(const v16bf*)(&cur[l16 * XP + kc * 32 + hi * 16]);
#pragma unroll
      for (int q = 0; q < 4; ++q) {
        const v16bf b = *(const AS1 v16bf*)(wr_t + (long)q * 16 * 768 + kc * 32);
        acc[q] = __builtin_amdgcn_wmma_f32_16x16x32_bf16(
            false, a, false, b, (short)0, acc[q], false, false);
      }
    }
#pragma unroll
    for (int kc = 0; kc < 8; ++kc) {  // h part
      v16bf a = *(const v16bf*)(&sh_h[l16 * HP + kc * 32 + hi * 16]);
#pragma unroll
      for (int q = 0; q < 4; ++q) {
        const v16bf b = *(const AS1 v16bf*)(wr_t + (long)q * 16 * 768 + 512 + kc * 32);
        acc[q] = __builtin_amdgcn_wmma_f32_16x16x32_bf16(
            false, a, false, b, (short)0, acc[q], false, false);
      }
    }

    // -------- lane-local LSTM cell update (rows m = v + 8*hi, col j) -------
    __bf16 hnew[8];
#pragma unroll
    for (int v = 0; v < 8; ++v) {
      int m = v + hi * 8;
      int len = sh_len[m];
      float iv = acc[0][v] + bI;
      float fv = acc[1][v] + bF;
      float gv = acc[2][v] + bG;
      float ov = acc[3][v] + bO;
      float cn = sigmoidf_(fv) * creg[v] + sigmoidf_(iv) * tanh_fast(gv);
      float hn = sigmoidf_(ov) * tanh_fast(cn);
      if (t < len) { creg[v] = cn; hreg[v] = hn; }
      hnew[v] = (__bf16)hreg[v];
    }
    __syncthreads();  // all WMMA reads of sh_h / sh_x done
#pragma unroll
    for (int v = 0; v < 8; ++v) {
      int m = v + hi * 8;
      sh_h[m * HP + j] = hnew[v];
    }
  }

#pragma unroll
  for (int v = 0; v < 8; ++v) {
    int m = v + hi * 8;
    out[(long)(row0 + m) * rowStride + colOff + j] = hreg[v];
  }
}

// ---------------------------------------------------------------------------
// Head: scores = rep . method3 ; softmax over BATCH dim ; method = attn-mix.
// Single block, 256 threads.
// ---------------------------------------------------------------------------
__global__ void attn_method_kernel(const float* __restrict__ rep,     // [64,512]
                                   const float* __restrict__ m3,      // [64,3,512]
                                   float* __restrict__ method) {      // [64,512]
  __shared__ float sS[3 * 64];
  __shared__ float sA[3 * 64];
  int tid = threadIdx.x;
  for (int task = tid; task < 192; task += 256) {
    int c = task / 64, b = task % 64;
    const float* r = rep + b * 512;
    const float* m = m3 + ((long)b * 3 + c) * 512;
    float s = 0.f;
    for (int d = 0; d < 512; ++d) s += r[d] * m[d];
    sS[c * 64 + b] = s;
  }
  __syncthreads();
  if (tid < 3) {
    int c = tid;
    float mx = -1e30f;
    for (int b = 0; b < 64; ++b) mx = fmaxf(mx, sS[c * 64 + b]);
    float sum = 0.f;
    for (int b = 0; b < 64; ++b) { float e = __expf(sS[c * 64 + b] - mx); sA[c * 64 + b] = e; sum += e; }
    float inv = 1.f / sum;
    for (int b = 0; b < 64; ++b) sA[c * 64 + b] *= inv;
  }
  __syncthreads();
  for (int idx = tid; idx < 64 * 512; idx += 256) {
    int b = idx / 512, d = idx % 512;
    float acc = 0.f;
    for (int c = 0; c < 3; ++c) acc += sA[c * 64 + b] * m3[((long)b * 3 + c) * 512 + d];
    method[idx] = acc;
  }
}

// ---------------------------------------------------------------------------
// Rel-method attention; algebraically reduced:
//  d1=rep.rel3, d2=method.rel3; a1=softmax_l(d1 masked); s2=sum_c a1*d2 masked
//  a2=softmax_l(s2) masked->0; expand = method * (cnt>0 ? sum_l a2 : 1)
// One block per batch row.
// ---------------------------------------------------------------------------
__global__ void rel_expand_kernel(const float* __restrict__ rep,
                                  const float* __restrict__ method,
                                  const float* __restrict__ rel3,    // [64,8,3,512]
                                  const int* __restrict__ rel_counts,
                                  float* __restrict__ expand) {      // [64,512]
  __shared__ float d1[24], d2[24], a1[24], s2[8], fac;
  int b = blockIdx.x;
  int tid = threadIdx.x, wave = tid >> 5, lane = tid & 31;
  int cnt = rel_counts[b];
  for (int it = 0; it < 6; ++it) {
    int task = wave + 8 * it;  // 48 tasks: (which, l, c)
    int which = task / 24, rem = task % 24;
    int l = rem / 3, c = rem % 3;
    const float* vec = rel3 + (((long)b * 8 + l) * 3 + c) * 512;
    const float* ref = (which ? method : rep) + (long)b * 512;
    float p = 0.f;
    for (int d = lane; d < 512; d += 32) p += ref[d] * vec[d];
    for (int off = 16; off > 0; off >>= 1) p += __shfl_down(p, off, 32);
    if (lane == 0) { if (which) d2[rem] = p; else d1[rem] = p; }
  }
  __syncthreads();
  if (tid < 3) {
    int c = tid;
    float mx = -1e30f;
    for (int l = 0; l < 8; ++l) {
      float s = (l < cnt) ? d1[l * 3 + c] : -1e9f;
      mx = fmaxf(mx, s);
    }
    float sum = 0.f;
    for (int l = 0; l < 8; ++l) {
      float s = (l < cnt) ? d1[l * 3 + c] : -1e9f;
      float e = __expf(s - mx);
      a1[l * 3 + c] = e;
      sum += e;
    }
    float inv = 1.f / sum;
    for (int l = 0; l < 8; ++l) a1[l * 3 + c] *= inv;
  }
  __syncthreads();
  if (tid < 8) {
    float s = -1e9f;
    if (tid < cnt) {
      s = 0.f;
      for (int c = 0; c < 3; ++c) s += a1[tid * 3 + c] * d2[tid * 3 + c];
    }
    s2[tid] = s;
  }
  __syncthreads();
  if (tid == 0) {
    if (cnt == 0) { fac = 1.0f; }
    else {
      float mx = -1e30f;
      for (int l = 0; l < 8; ++l) mx = fmaxf(mx, s2[l]);
      float sum = 0.f, vsum = 0.f;
      float e[8];
      for (int l = 0; l < 8; ++l) { e[l] = __expf(s2[l] - mx); sum += e[l]; }
      for (int l = 0; l < cnt; ++l) vsum += e[l];
      fac = vsum / sum;
    }
  }
  __syncthreads();
  for (int d = tid; d < 512; d += blockDim.x)
    expand[(long)b * 512 + d] = method[(long)b * 512 + d] * fac;
}

// ---------------------------------------------------------------------------
// GRU cell: out = gru(x=method, h=expand).  One thread per (b, j).
// ---------------------------------------------------------------------------
__global__ void gru_kernel(const float* __restrict__ x, const float* __restrict__ h,
                           const float* __restrict__ w_ih, const float* __restrict__ w_hh,
                           const float* __restrict__ b_ih, const float* __restrict__ b_hh,
                           float* __restrict__ out) {
  int id = blockIdx.x * blockDim.x + threadIdx.x;
  if (id >= 64 * 512) return;
  int b = id / 512, j = id % 512;
  const float* xb = x + (long)b * 512;
  const float* hb = h + (long)b * 512;
  float ir = b_ih[j], iz = b_ih[512 + j], in_ = b_ih[1024 + j];
  float hr = b_hh[j], hz = b_hh[512 + j], hn = b_hh[1024 + j];
  const float* wr = w_ih + (long)j * 512;
  const float* wz = w_ih + (long)(512 + j) * 512;
  const float* wn = w_ih + (long)(1024 + j) * 512;
  const float* vr = w_hh + (long)j * 512;
  const float* vz = w_hh + (long)(512 + j) * 512;
  const float* vn = w_hh + (long)(1024 + j) * 512;
  for (int k = 0; k < 512; ++k) {
    float xv = xb[k], hv = hb[k];
    ir += xv * wr[k]; iz += xv * wz[k]; in_ += xv * wn[k];
    hr += hv * vr[k]; hz += hv * vz[k]; hn += hv * vn[k];
  }
  float r = sigmoidf_(ir + hr);
  float z = sigmoidf_(iz + hz);
  float n = tanh_fast(in_ + r * hn);
  out[id] = (1.0f - z) * n + z * hb[j];
}

// ---------------------------------------------------------------------------
// FC:  out[b][j] = bias[j] + A[b]·W[j][0:Ka] + B2[b]·W[j][Ka:Ka+Kb]
// ---------------------------------------------------------------------------
__global__ void fc_kernel(const float* __restrict__ A, int Ka,
                          const float* __restrict__ B2, int Kb,
                          const float* __restrict__ W, const float* __restrict__ bias,
                          float* __restrict__ out, int NB, int J) {
  int id = blockIdx.x * blockDim.x + threadIdx.x;
  if (id >= NB * J) return;
  int b = id / J, j = id % J;
  const float* wr = W + (long)j * (Ka + Kb);
  float s = bias[j];
  const float* a = A + (long)b * Ka;
  for (int k = 0; k < Ka; ++k) s += a[k] * wr[k];
  if (B2) {
    const float* bb = B2 + (long)b * Kb;
    for (int k = 0; k < Kb; ++k) s += bb[k] * wr[Ka + k];
  }
  out[id] = s;
}

// ---------------------------------------------------------------------------
// BatchNorm (training stats over batch of 64, biased var) + ReLU.  One block.
// ---------------------------------------------------------------------------
__global__ void bn_relu_kernel(const float* __restrict__ x, float* __restrict__ y,
                               const float* __restrict__ g, const float* __restrict__ bt,
                               int cols) {
  for (int j = threadIdx.x; j < cols; j += blockDim.x) {
    float s = 0.f, s2 = 0.f;
    for (int b = 0; b < 64; ++b) { float v = x[(long)b * cols + j]; s += v; s2 += v * v; }
    float mu = s * (1.0f / 64.0f);
    float var = s2 * (1.0f / 64.0f) - mu * mu;
    float inv = rsqrtf(var + 1e-5f);
    float gg = g[j], bb = bt[j];
    for (int b = 0; b < 64; ++b) {
      float v = gg * (x[(long)b * cols + j] - mu) * inv + bb;
      y[(long)b * cols + j] = fmaxf(v, 0.0f);
    }
  }
}

// ---------------------------------------------------------------------------
// Host launcher
// ---------------------------------------------------------------------------
extern "C" void kernel_launch(void* const* d_in, const int* in_sizes, int n_in,
                              void* d_out, int out_size, void* d_ws, size_t ws_size,
                              hipStream_t stream) {
  (void)in_sizes; (void)n_in; (void)out_size; (void)ws_size;
  const float* word_emb = (const float*)d_in[0];
  const float* code_emb = (const float*)d_in[1];
  // encoder bases: report=2, token=8, api=14, comment=20 (w_ih_f,w_hh_f,b_f,w_ih_b,w_hh_b,b_b)
  const int encBase[4] = {2, 8, 14, 20};
  const float* gru_wih = (const float*)d_in[26];
  const float* gru_whh = (const float*)d_in[27];
  const float* gru_bih = (const float*)d_in[28];
  const float* gru_bhh = (const float*)d_in[29];
  const float* sim_w1 = (const float*)d_in[30]; const float* sim_b1 = (const float*)d_in[31];
  const float* sim_g1 = (const float*)d_in[32]; const float* sim_t1 = (const float*)d_in[33];
  const float* sim_w2 = (const float*)d_in[34]; const float* sim_b2 = (const float*)d_in[35];
  const float* sim_g2 = (const float*)d_in[36]; const float* sim_t2 = (const float*)d_in[37];
  const float* sim_w3 = (const float*)d_in[38]; const float* sim_b3 = (const float*)d_in[39];
  const int* report = (const int*)d_in[40]; const int* report_len = (const int*)d_in[41];
  const int* token  = (const int*)d_in[42]; const int* token_len  = (const int*)d_in[43];
  const int* api    = (const int*)d_in[44]; const int* api_len    = (const int*)d_in[45];
  const int* commnt = (const int*)d_in[46]; const int* commnt_len = (const int*)d_in[47];
  const int* rtok = (const int*)d_in[48]; const int* rtok_len = (const int*)d_in[49];
  const int* rapi = (const int*)d_in[50]; const int* rapi_len = (const int*)d_in[51];
  const int* rcom = (const int*)d_in[52]; const int* rcom_len = (const int*)d_in[53];
  const int* rel_counts = (const int*)d_in[54];

  // ---- workspace carve-up -------------------------------------------------
  char* ws = (char*)d_ws;
  size_t cur = 0;
  auto carve = [&](size_t bytes) -> char* {
    char* p = ws + cur;
    cur += (bytes + 255) & ~(size_t)255;
    return p;
  };
  __bf16* wcat[8];  // rep_f, rep_b, tok_f, tok_b, api_f, api_b, com_f, com_b
  for (int i = 0; i < 8; ++i) wcat[i] = (__bf16*)carve((size_t)1024 * 768 * 2);

  __bf16* Xrep = (__bf16*)carve((size_t)64 * 128 * 512 * 2);
  __bf16* Xtok = (__bf16*)carve((size_t)64 * 128 * 512 * 2);
  __bf16* Xapi = (__bf16*)carve((size_t)64 * 32 * 512 * 2);
  __bf16* Xcom = (__bf16*)carve((size_t)64 * 64 * 512 * 2);
  __bf16* Xrt  = (__bf16*)carve((size_t)512 * 64 * 512 * 2);
  __bf16* Xra  = (__bf16*)carve((size_t)512 * 16 * 512 * 2);
  __bf16* Xrc  = (__bf16*)carve((size_t)512 * 32 * 512 * 2);

  float* repB    = (float*)carve((size_t)64 * 512 * 4);
  float* m3B     = (float*)carve((size_t)64 * 3 * 512 * 4);
  float* rel3B   = (float*)carve((size_t)512 * 3 * 512 * 4);
  float* methodB = (float*)carve((size_t)64 * 512 * 4);
  float* expandB = (float*)carve((size_t)64 * 512 * 4);
  float* augB    = (float*)carve((size_t)64 * 512 * 4);
  float* h1B     = (float*)carve((size_t)64 * 512 * 4);
  float* h1rB    = (float*)carve((size_t)64 * 512 * 4);
  float* h2B     = (float*)carve((size_t)64 * 128 * 4);
  float* h2rB    = (float*)carve((size_t)64 * 128 * 4);

  // ---- 1) weight concat + convert ----------------------------------------
  {
    dim3 g((1024 * 768 + 255) / 256), b(256);
    for (int e = 0; e < 4; ++e) {
      int base = encBase[e];
      build_wcat_kernel<<<g, b, 0, stream>>>((const float*)d_in[base + 0],
                                             (const float*)d_in[base + 1], wcat[e * 2 + 0]);
      build_wcat_kernel<<<g, b, 0, stream>>>((const float*)d_in[base + 3],
                                             (const float*)d_in[base + 4], wcat[e * 2 + 1]);
    }
  }

  // ---- 2) embedding gathers (f32 -> bf16) ---------------------------------
  struct GatherT { const float* emb; const int* ids; __bf16* X; long n; };
  GatherT gts[7] = {
      {word_emb, report, Xrep, 64L * 128},
      {code_emb, token,  Xtok, 64L * 128},
      {code_emb, api,    Xapi, 64L * 32},
      {word_emb, commnt, Xcom, 64L * 64},
      {code_emb, rtok,   Xrt,  512L * 64},
      {code_emb, rapi,   Xra,  512L * 16},
      {word_emb, rcom,   Xrc,  512L * 32},
  };
  for (int i = 0; i < 7; ++i) {
    long total = gts[i].n * 128;
    dim3 g((unsigned)((total + 255) / 256)), b(256);
    gather_embed_kernel<<<g, b, 0, stream>>>(gts[i].emb, gts[i].ids, gts[i].X, gts[i].n);
  }

  // ---- 3) all 14 LSTM recurrences (independent, launched back to back) ----
  struct RecT { const __bf16* X; const int* lens; int wslot; int N, T;
                float* out; int rowStride, colOff; };
  RecT recs[7] = {
      {Xrep, report_len, 0, 64, 128, repB, 512, 0},       // enc_report
      {Xtok, token_len,  2, 64, 128, m3B, 1536, 0},       // enc_token  -> c=0
      {Xapi, api_len,    4, 64, 32,  m3B, 1536, 512},     // enc_api    -> c=1
      {Xcom, commnt_len, 6, 64, 64,  m3B, 1536, 1024},    // enc_comment-> c=2
      {Xrt,  rtok_len,   2, 512, 64, rel3B, 1536, 0},
      {Xra,  rapi_len,   4, 512, 16, rel3B, 1536, 512},
      {Xrc,  rcom_len,   6, 512, 32, rel3B, 1536, 1024},
  };
  for (int i = 0; i < 7; ++i) {
    const RecT& r = recs[i];
    int encB = encBase[r.wslot / 2];
    // forward direction
    lstm_rec_kernel<<<dim3(r.N / 16), dim3(512), 0, stream>>>(
        r.X, r.lens, wcat[r.wslot], (const float*)d_in[encB + 2],
        r.out, r.T, r.rowStride, r.colOff + 0, 0);
    // backward direction (reads x at reversed-within-length positions)
    lstm_rec_kernel<<<dim3(r.N / 16), dim3(512), 0, stream>>>(
        r.X, r.lens, wcat[r.wslot + 1], (const float*)d_in[encB + 5],
        r.out, r.T, r.rowStride, r.colOff + 256, 1);
  }

  // ---- 4) head ------------------------------------------------------------
  attn_method_kernel<<<dim3(1), dim3(256), 0, stream>>>(repB, m3B, methodB);
  rel_expand_kernel<<<dim3(64), dim3(256), 0, stream>>>(repB, methodB, rel3B,
                                                        rel_counts, expandB);
  gru_kernel<<<dim3((64 * 512 + 255) / 256), dim3(256), 0, stream>>>(
      methodB, expandB, gru_wih, gru_whh, gru_bih, gru_bhh, augB);

  // sim MLP: feat = [aug | rep] -> 512 -> BN/ReLU -> 128 -> BN/ReLU -> 1
  fc_kernel<<<dim3((64 * 512 + 255) / 256), dim3(256), 0, stream>>>(
      augB, 512, repB, 512, sim_w1, sim_b1, h1B, 64, 512);
  bn_relu_kernel<<<dim3(1), dim3(512), 0, stream>>>(h1B, h1rB, sim_g1, sim_t1, 512);
  fc_kernel<<<dim3((64 * 128 + 255) / 256), dim3(256), 0, stream>>>(
      h1rB, 512, nullptr, 0, sim_w2, sim_b2, h2B, 64, 128);
  bn_relu_kernel<<<dim3(1), dim3(256), 0, stream>>>(h2B, h2rB, sim_g2, sim_t2, 128);
  fc_kernel<<<dim3(1), dim3(64), 0, stream>>>(
      h2rB, 128, nullptr, 0, sim_w3, sim_b3, (float*)d_out, 64, 1);
}
